// EMGPatchTransformer_56736517980830
// MI455X (gfx1250) — matrix operand order repaired and checked by reading, not compiled, not run on hardware
//
#include <hip/hip_runtime.h>
#include <math.h>

// ---------------- problem constants (from reference) ----------------
constexpr int BATCH = 32;
constexpr int CCH   = 8;      // emg channels
constexpr int TT    = 20000;  // time
constexpr int PP    = 25;     // patch
constexpr int LL    = TT / PP;            // 800 tokens
constexpr int NTOK  = BATCH * LL;         // 25600
constexpr int DD    = 512;
constexpr int HH    = 8;
constexpr int DHd   = DD / HH;            // 64
constexpr int MM    = 256;                // random features
constexpr int NLL   = 2;
constexpr int FFNn  = 1024;
constexpr int NCn   = 10;
constexpr int HIDn  = 128;
constexpr int CP    = CCH * PP;           // 200

// ---------------- WMMA types ----------------
typedef __bf16 v16bf __attribute__((ext_vector_type(16)));
typedef float  v8f   __attribute__((ext_vector_type(8)));

union Frag  { uint4 q[2]; v16bf v; };     // 16 bf16 = 32B
union Half8 { uint4 q;    __bf16 h[8]; }; // 8 bf16 = 16B

// GEMM tiling: 128x64 block tile, K-step 32, 8 waves (wave32), each wave = 16x64
constexpr int GBM = 128;
constexpr int GBN = 64;
constexpr int GBK = 32;
constexpr int LSA = 40;   // LDS row stride (bf16 elems) -> 80B, keeps 16B alignment
constexpr int LSB = 40;

// epilogues: 0=none 1=+bias 2=gelu(x+bias) 3=tanh(x+bias) 4=x+bias+res 5=exp(x-rowsub)*oscale
__global__ __launch_bounds__(256) void gemm_kernel(
    const float* __restrict__ A, const float* __restrict__ Bm, float* __restrict__ C,
    const float* __restrict__ bias, const float* __restrict__ res,
    const float* __restrict__ rowsub,
    int Mi, int Ni, int Ki, int lda, int ldb, int ldc,
    long zA, long zB, long zC, long zR,
    float alpha, float oscale, int epi, int transA)
{
  __shared__ __align__(16) __bf16 sA[GBM * LSA];
  __shared__ __align__(16) __bf16 sB[GBN * LSB];

  const int tid = threadIdx.x;
  const int z   = blockIdx.z;
  A  += (long)z * zA;
  Bm += (long)z * zB;
  C  += (long)z * zC;
  const float* rp = res    ? res    + (long)z * zC : nullptr;
  const float* rs = rowsub ? rowsub + (long)z * zR : nullptr;

  const int m0 = blockIdx.y * GBM;
  const int n0 = blockIdx.x * GBN;
  const int w    = tid >> 5;
  const int lane = tid & 31;
  const int lr   = lane & 15;
  const int lh   = lane >> 4;

  v8f acc[4];
#pragma unroll
  for (int j = 0; j < 4; ++j)
#pragma unroll
    for (int e = 0; e < 8; ++e) acc[j][e] = 0.f;

  const int ar = tid >> 1;          // A tile row 0..127
  const int ak = (tid & 1) * 16;    // 16 consecutive k per thread
  const int atk  = tid & 31;        // transA: k within tile
  const int atm  = (tid >> 5) * 16; // transA: 16 contiguous m per thread
  const int bn = tid & 63;          // B tile col
  const int bk = (tid >> 6) * 8;    // 8 consecutive k per thread

  const bool fullM = (m0 + GBM) <= Mi;
  const bool fullN = (n0 + GBN) <= Ni;

  for (int k0 = 0; k0 < Ki; k0 += GBK) {
    const bool fullK = (k0 + GBK) <= Ki;

    // ---------------- stage A tile (f32 -> bf16, alpha folded in) ----------------
    if (fullM && fullK && !transA) {
      // vector path: 4x float4 global loads, 2x uint4 LDS stores
      const float4* src = (const float4*)&A[(long)(m0 + ar) * lda + k0 + ak];
      Frag f;
#pragma unroll
      for (int q = 0; q < 4; ++q) {
        float4 v = src[q];
        f.v[4 * q + 0] = (__bf16)(v.x * alpha);
        f.v[4 * q + 1] = (__bf16)(v.y * alpha);
        f.v[4 * q + 2] = (__bf16)(v.z * alpha);
        f.v[4 * q + 3] = (__bf16)(v.w * alpha);
      }
      uint4* d = (uint4*)&sA[ar * LSA + ak];
      d[0] = f.q[0];
      d[1] = f.q[1];
      if (k0 + GBK < Ki) __builtin_prefetch(src + 8, 0, 1);
    } else if (fullM && fullK && transA) {
      // coalesced along M, transposed LDS stores
      const float4* src = (const float4*)&A[(long)(k0 + atk) * lda + m0 + atm];
#pragma unroll
      for (int q = 0; q < 4; ++q) {
        float4 v = src[q];
        sA[(atm + 4 * q + 0) * LSA + atk] = (__bf16)(v.x * alpha);
        sA[(atm + 4 * q + 1) * LSA + atk] = (__bf16)(v.y * alpha);
        sA[(atm + 4 * q + 2) * LSA + atk] = (__bf16)(v.z * alpha);
        sA[(atm + 4 * q + 3) * LSA + atk] = (__bf16)(v.w * alpha);
      }
    } else {
      // guarded tail path
      int gm = m0 + ar;
#pragma unroll
      for (int i = 0; i < 16; ++i) {
        int gk = k0 + ak + i;
        float v = 0.f;
        if (gm < Mi && gk < Ki)
          v = transA ? A[(long)gk * lda + gm] : A[(long)gm * lda + gk];
        sA[ar * LSA + ak + i] = (__bf16)(v * alpha);
      }
    }

    // ---------------- stage B tile transposed: sB[n][k] ----------------
    if (fullN && fullK) {
      int gn = n0 + bn;
      Half8 f;
#pragma unroll
      for (int i = 0; i < 8; ++i)
        f.h[i] = (__bf16)Bm[(long)(k0 + bk + i) * ldb + gn];
      *(uint4*)&sB[bn * LSB + bk] = f.q;
    } else {
      int gn = n0 + bn;
#pragma unroll
      for (int i = 0; i < 8; ++i) {
        int gk = k0 + bk + i;
        float v = (gn < Ni && gk < Ki) ? Bm[(long)gk * ldb + gn] : 0.f;
        sB[bn * LSB + bk + i] = (__bf16)v;
      }
    }
    __syncthreads();

    // A frag: row = lr, elems 0..7 = K 8*lh.., elems 8..15 = K 16+8*lh..
    Frag af;
    const __bf16* arow = &sA[(w * 16 + lr) * LSA];
    af.q[0] = *(const uint4*)(arow + 8 * lh);
    af.q[1] = *(const uint4*)(arow + 16 + 8 * lh);
#pragma unroll
    for (int j = 0; j < 4; ++j) {
      // B frag: col = lr (of 16-col tile j), elems 0..7 = K 8*lh.., elems 8..15 = K 16+8*lh..
      Frag bf;
      const __bf16* bcol = &sB[(j * 16 + lr) * LSB + 8 * lh];
      bf.q[0] = *(const uint4*)(bcol);
      bf.q[1] = *(const uint4*)(bcol + 16);
      acc[j] = __builtin_amdgcn_wmma_f32_16x16x32_bf16(
          false, af.v, false, bf.v, (short)0, acc[j], false, false);
    }
    __syncthreads();
  }

  // epilogue: C row = m0 + w*16 + 8*lh + e, col = n0 + j*16 + lr
#pragma unroll
  for (int j = 0; j < 4; ++j) {
    int gn = n0 + j * 16 + lr;
    if (gn >= Ni) continue;
    float bv = bias ? bias[gn] : 0.f;
#pragma unroll
    for (int e = 0; e < 8; ++e) {
      int gm = m0 + w * 16 + 8 * lh + e;
      if (gm >= Mi) continue;
      float xv = acc[j][e];
      float o;
      if (epi == 0)      o = xv;
      else if (epi == 1) o = xv + bv;
      else if (epi == 2) { float t = xv + bv; o = 0.5f * t * (1.f + erff(t * 0.70710678118f)); }
      else if (epi == 3) o = tanhf(xv + bv);
      else if (epi == 4) o = xv + bv + rp[(long)gm * ldc + gn];
      else               o = __expf(xv - rs[gm]) * oscale;
      C[(long)gm * ldc + gn] = o;
    }
  }
}

// ---------------- helper kernels (wave32) ----------------
__device__ __forceinline__ float wave_sum(float s) {
#pragma unroll
  for (int o = 16; o; o >>= 1) s += __shfl_xor(s, o, 32);
  return s;
}

__global__ __launch_bounds__(256) void patchify_kernel(const float* __restrict__ x,
                                                       float* __restrict__ tok) {
  long i = (long)blockIdx.x * 256 + threadIdx.x;   // over NTOK*CP
  if (i >= (long)NTOK * CP) return;
  int cp = (int)(i % CP);
  long t = i / CP;
  int c = cp / PP, p = cp % PP;
  int b = (int)(t / LL), l = (int)(t % LL);
  tok[i] = x[((long)b * CCH + c) * TT + (long)l * PP + p];
}

__global__ __launch_bounds__(256) void ln_kernel(const float* __restrict__ in,
                                                 float* __restrict__ out,
                                                 const float* __restrict__ g,
                                                 const float* __restrict__ bta,
                                                 int rows, int cols) {
  int row  = blockIdx.x * 8 + (threadIdx.x >> 5);
  int lane = threadIdx.x & 31;
  if (row >= rows) return;
  const float* r = in + (long)row * cols;
  float s = 0.f, s2 = 0.f;
  for (int c = lane; c < cols; c += 32) { float v = r[c]; s += v; s2 += v * v; }
  s = wave_sum(s); s2 = wave_sum(s2);
  float mu = s / cols;
  float var = s2 / cols - mu * mu;
  float rstd = rsqrtf(var + 1e-5f);
  float* o = out + (long)row * cols;
  for (int c = lane; c < cols; c += 32) o[c] = (r[c] - mu) * rstd * g[c] + bta[c];
}

// nsq[(b*H+hh)*L + l] = coef * sum_d q[t, hh*64+d]^2
__global__ __launch_bounds__(256) void nsq_kernel(const float* __restrict__ Q,
                                                  float* __restrict__ nsq, float coef) {
  int idx  = blockIdx.x * 8 + (threadIdx.x >> 5);   // over NTOK*HH
  if (idx >= NTOK * HH) return;
  int hh = idx % HH, t = idx / HH;
  int lane = threadIdx.x & 31;
  const float* q = Q + (long)t * DD + hh * DHd;
  float s = q[lane] * q[lane] + q[lane + 32] * q[lane + 32];
  s = wave_sum(s);
  if (lane == 0) {
    int b = t / LL, l = t % LL;
    nsq[((long)b * HH + hh) * LL + l] = coef * s;
  }
}

// Ksum[hh][m] = sum_l Kf[hh][l][m]
__global__ __launch_bounds__(256) void ksum_kernel(const float* __restrict__ Kf,
                                                   float* __restrict__ Ksum) {
  int i = blockIdx.x * 256 + threadIdx.x;   // HH*MM
  if (i >= HH * MM) return;
  int hh = i >> 8, m = i & 255;
  const float* p = Kf + ((long)hh * LL) * MM + m;
  float s = 0.f;
  for (int l = 0; l < LL; ++l) s += p[(long)l * MM];
  Ksum[i] = s;
}

// norm = Qf[hh,l,:].Ksum[hh,:]; dst[l, hh*64+d] = Oatt[hh,l,d] / max(norm,1e-6)
__global__ __launch_bounds__(256) void merge_kernel(const float* __restrict__ Qf,
                                                    const float* __restrict__ Ksum,
                                                    const float* __restrict__ Oatt,
                                                    float* __restrict__ dst) {
  int idx  = blockIdx.x * 8 + (threadIdx.x >> 5);   // over HH*LL
  if (idx >= HH * LL) return;
  int hh = idx / LL, l = idx % LL;
  int lane = threadIdx.x & 31;
  const float* qf = Qf + ((long)hh * LL + l) * MM;
  const float* ks = Ksum + hh * MM;
  float s = 0.f;
#pragma unroll
  for (int i = 0; i < 8; ++i) s += qf[lane + 32 * i] * ks[lane + 32 * i];
  s = wave_sum(s);
  float inv = 1.f / fmaxf(s, 1e-6f);
  const float* o = Oatt + ((long)hh * LL + l) * DHd;
  float* d = dst + (long)l * DD + hh * DHd;
  d[lane]      = o[lane] * inv;
  d[lane + 32] = o[lane + 32] * inv;
}

__global__ __launch_bounds__(256) void score_kernel(const float* __restrict__ s1,
                                                    const float* __restrict__ w2,
                                                    float* __restrict__ sc) {
  int t = blockIdx.x * 8 + (threadIdx.x >> 5);
  if (t >= NTOK) return;
  int lane = threadIdx.x & 31;
  const float* r = s1 + (long)t * HIDn;
  float s = r[lane] * w2[lane] + r[lane + 32] * w2[lane + 32] +
            r[lane + 64] * w2[lane + 64] + r[lane + 96] * w2[lane + 96];
  s = wave_sum(s);
  if (lane == 0) sc[t] = s;
}

// attentive stats pooling: one block (512 thr) per batch
__global__ __launch_bounds__(512) void pool_kernel(const float* __restrict__ hbuf,
                                                   const float* __restrict__ sc,
                                                   float* __restrict__ feat) {
  __shared__ float ex[LL];
  __shared__ float red[16];
  __shared__ float stat[2];
  int b = blockIdx.x, tid = threadIdx.x;
  const float* s = sc + (long)b * LL;
  float m = -3.4e38f;
  for (int l = tid; l < LL; l += 512) m = fmaxf(m, s[l]);
#pragma unroll
  for (int o = 16; o; o >>= 1) m = fmaxf(m, __shfl_xor(m, o, 32));
  if ((tid & 31) == 0) red[tid >> 5] = m;
  __syncthreads();
  if (tid == 0) { float v = red[0]; for (int i = 1; i < 16; ++i) v = fmaxf(v, red[i]); stat[0] = v; }
  __syncthreads();
  float mx = stat[0];
  float ls = 0.f;
  for (int l = tid; l < LL; l += 512) { float e = __expf(s[l] - mx); ex[l] = e; ls += e; }
  ls = wave_sum(ls);
  if ((tid & 31) == 0) red[tid >> 5] = ls;
  __syncthreads();
  if (tid == 0) { float v = 0.f; for (int i = 0; i < 16; ++i) v += red[i]; stat[1] = v; }
  __syncthreads();
  float inv = 1.f / stat[1];
  int d = tid;  // 0..511 == DD
  const float* hb = hbuf + (long)b * LL * DD + d;
  float mu = 0.f, m2 = 0.f;
  for (int l = 0; l < LL; ++l) {
    float v = hb[(long)l * DD];
    float a = ex[l];
    mu += a * v; m2 += a * v * v;
  }
  mu *= inv; m2 *= inv;
  float var = m2 - mu * mu;
  feat[(long)b * (2 * DD) + d]      = mu;
  feat[(long)b * (2 * DD) + DD + d] = sqrtf(fmaxf(var, 1e-8f));
}

__global__ __launch_bounds__(256) void head_kernel(const float* __restrict__ feat,
                                                   const float* __restrict__ W,
                                                   const float* __restrict__ bb,
                                                   float* __restrict__ out) {
  int idx = blockIdx.x * 8 + (threadIdx.x >> 5);    // BATCH*NCn
  if (idx >= BATCH * NCn) return;
  int b = idx / NCn, c = idx % NCn;
  int lane = threadIdx.x & 31;
  const float* f = feat + (long)b * (2 * DD);
  float s = 0.f;
  for (int k = lane; k < 2 * DD; k += 32) s += f[k] * W[(long)k * NCn + c];
  s = wave_sum(s);
  if (lane == 0) out[idx] = s + bb[c];
}

// ---------------- host orchestration ----------------
extern "C" void kernel_launch(void* const* d_in, const int* in_sizes, int n_in,
                              void* d_out, int out_size, void* d_ws, size_t ws_size,
                              hipStream_t stream) {
  (void)in_sizes; (void)n_in; (void)out_size; (void)ws_size;
  const float* x       = (const float*)d_in[0];
  const float* patchW  = (const float*)d_in[1];
  const float* patchb  = (const float*)d_in[2];
  const float* patchg  = (const float*)d_in[3];
  const float* patchbe = (const float*)d_in[4];
  const float* ln1g    = (const float*)d_in[5];
  const float* ln1b    = (const float*)d_in[6];
  const float* qW      = (const float*)d_in[7];
  const float* kW      = (const float*)d_in[8];
  const float* vW      = (const float*)d_in[9];
  const float* oW      = (const float*)d_in[10];
  const float* ob      = (const float*)d_in[11];
  const float* omega   = (const float*)d_in[12];
  const float* ln2g    = (const float*)d_in[13];
  const float* ln2b    = (const float*)d_in[14];
  const float* f1W     = (const float*)d_in[15];
  const float* f1b     = (const float*)d_in[16];
  const float* f2W     = (const float*)d_in[17];
  const float* f2b     = (const float*)d_in[18];
  const float* poolW1  = (const float*)d_in[19];
  const float* poolW2  = (const float*)d_in[20];
  const float* headW   = (const float*)d_in[21];
  const float* headb   = (const float*)d_in[22];
  float* out = (float*)d_out;

  // workspace carve-out (all sizes multiple of 256B)
  size_t off = 0;
  auto take = [&](size_t bytes) {
    void* p = (char*)d_ws + off;
    off += (bytes + 255) & ~(size_t)255;
    return p;
  };
  float* hbuf = (float*)take((size_t)NTOK * DD * 4);
  float* xln  = (float*)take((size_t)NTOK * DD * 4);
  float* Qb   = (float*)take((size_t)NTOK * DD * 4);
  float* Kb   = (float*)take((size_t)NTOK * DD * 4);
  float* Vb   = (float*)take((size_t)NTOK * DD * 4);
  float* mrg  = (float*)take((size_t)NTOK * DD * 4);
  float* Qf   = (float*)take((size_t)HH * LL * MM * 4);
  float* Kf   = (float*)take((size_t)HH * LL * MM * 4);
  float* KVb  = (float*)take((size_t)HH * MM * DHd * 4);
  float* Oatt = (float*)take((size_t)HH * LL * DHd * 4);
  float* nsqQ = (float*)take((size_t)BATCH * HH * LL * 4);
  float* nsqK = (float*)take((size_t)BATCH * HH * LL * 4);
  float* Ksum = (float*)take((size_t)HH * MM * 4);
  float* scb  = (float*)take((size_t)NTOK * 4);
  float* feat = (float*)take((size_t)BATCH * 2 * DD * 4);
  // aliases (lifetimes disjoint): tok before layers; ffn-mid spans Qb+Kb; s1 after layers
  float* tok    = Qb;
  float* ffnmid = Qb;
  float* s1     = Vb;

  auto gemm = [&](const float* A, const float* Bp, float* Cp,
                  const float* bias, const float* res, const float* rowsub,
                  int Mi, int Ni, int Ki, int lda, int ldb, int ldc,
                  long zA, long zB, long zC, long zR, int nz,
                  float alpha, float oscale, int epi, int transA) {
    dim3 g((Ni + GBN - 1) / GBN, (Mi + GBM - 1) / GBM, nz);
    gemm_kernel<<<g, dim3(256), 0, stream>>>(A, Bp, Cp, bias, res, rowsub,
                                             Mi, Ni, Ki, lda, ldb, ldc,
                                             zA, zB, zC, zR, alpha, oscale, epi, transA);
  };

  const float phiAlpha = 0.35355339059327373f;  // 1 / 64^(1/4)
  const float nsqCoef  = 0.0625f;               // 0.5 / sqrt(64)
  const float phiScale = 0.0625f;               // 1 / sqrt(256)

  // ---- patchify + embed + LN ----
  {
    long n = (long)NTOK * CP;
    patchify_kernel<<<dim3((unsigned)((n + 255) / 256)), dim3(256), 0, stream>>>(x, tok);
    gemm(tok, patchW, xln, patchb, nullptr, nullptr,
         NTOK, DD, CP, CP, DD, DD, 0, 0, 0, 0, 1, 1.f, 1.f, /*bias*/1, 0);
    ln_kernel<<<dim3(NTOK / 8), dim3(256), 0, stream>>>(xln, hbuf, patchg, patchbe, NTOK, DD);
  }

  // ---- transformer layers ----
  for (int l = 0; l < NLL; ++l) {
    const float* qWl = qW + (long)l * DD * DD;
    const float* kWl = kW + (long)l * DD * DD;
    const float* vWl = vW + (long)l * DD * DD;
    const float* oWl = oW + (long)l * DD * DD;
    const float* obl = ob + (long)l * DD;
    const float* oml = omega + (long)l * DHd * MM;

    ln_kernel<<<dim3(NTOK / 8), dim3(256), 0, stream>>>(hbuf, xln, ln1g + (long)l * DD,
                                                        ln1b + (long)l * DD, NTOK, DD);
    gemm(xln, qWl, Qb, nullptr, nullptr, nullptr, NTOK, DD, DD, DD, DD, DD,
         0, 0, 0, 0, 1, 1.f, 1.f, 0, 0);
    gemm(xln, kWl, Kb, nullptr, nullptr, nullptr, NTOK, DD, DD, DD, DD, DD,
         0, 0, 0, 0, 1, 1.f, 1.f, 0, 0);
    gemm(xln, vWl, Vb, nullptr, nullptr, nullptr, NTOK, DD, DD, DD, DD, DD,
         0, 0, 0, 0, 1, 1.f, 1.f, 0, 0);
    nsq_kernel<<<dim3(NTOK * HH / 8), dim3(256), 0, stream>>>(Qb, nsqQ, nsqCoef);
    nsq_kernel<<<dim3(NTOK * HH / 8), dim3(256), 0, stream>>>(Kb, nsqK, nsqCoef);

    for (int b = 0; b < BATCH; ++b) {
      const float* Qbb = Qb + (long)b * LL * DD;
      const float* Kbb = Kb + (long)b * LL * DD;
      const float* Vbb = Vb + (long)b * LL * DD;
      // phi: Qf[hh,l,m] = exp(alpha*Q . omega - nsq) / sqrt(M)   (z = head)
      gemm(Qbb, oml, Qf, nullptr, nullptr, nsqQ + (long)b * HH * LL,
           LL, MM, DHd, DD, MM, MM,
           /*zA*/DHd, /*zB*/0, /*zC*/(long)LL * MM, /*zR*/LL, HH,
           phiAlpha, phiScale, /*phi*/5, 0);
      gemm(Kbb, oml, Kf, nullptr, nullptr, nsqK + (long)b * HH * LL,
           LL, MM, DHd, DD, MM, MM,
           DHd, 0, (long)LL * MM, LL, HH,
           phiAlpha, phiScale, 5, 0);
      ksum_kernel<<<dim3(HH * MM / 256), dim3(256), 0, stream>>>(Kf, Ksum);
      // KV[hh] = Kf[hh]^T (L x M)^T @ V[hh] (L x 64)  -> (M x 64)
      gemm(Kf, Vbb, KVb, nullptr, nullptr, nullptr,
           MM, DHd, LL, /*lda*/MM, /*ldb*/DD, /*ldc*/DHd,
           (long)LL * MM, /*zB*/DHd, (long)MM * DHd, 0, HH,
           1.f, 1.f, 0, /*transA*/1);
      // Oatt[hh] = Qf[hh] (L x M) @ KV[hh] (M x 64)
      gemm(Qf, KVb, Oatt, nullptr, nullptr, nullptr,
           LL, DHd, MM, MM, DHd, DHd,
           (long)LL * MM, (long)MM * DHd, (long)LL * DHd, 0, HH,
           1.f, 1.f, 0, 0);
      merge_kernel<<<dim3(HH * LL / 8), dim3(256), 0, stream>>>(Qf, Ksum, Oatt,
                                                                mrg + (long)b * LL * DD);
    }
    // h += merged @ oW + ob  (in-place residual)
    gemm(mrg, oWl, hbuf, obl, hbuf, nullptr, NTOK, DD, DD, DD, DD, DD,
         0, 0, 0, 0, 1, 1.f, 1.f, /*bias+res*/4, 0);

    // FFN
    ln_kernel<<<dim3(NTOK / 8), dim3(256), 0, stream>>>(hbuf, xln, ln2g + (long)l * DD,
                                                        ln2b + (long)l * DD, NTOK, DD);
    gemm(xln, f1W + (long)l * DD * FFNn, ffnmid, f1b + (long)l * FFNn, nullptr, nullptr,
         NTOK, FFNn, DD, DD, FFNn, FFNn, 0, 0, 0, 0, 1, 1.f, 1.f, /*gelu*/2, 0);
    gemm(ffnmid, f2W + (long)l * FFNn * DD, hbuf, f2b + (long)l * DD, hbuf, nullptr,
         NTOK, DD, FFNn, FFNn, DD, DD, 0, 0, 0, 0, 1, 1.f, 1.f, /*bias+res*/4, 0);
  }

  // ---- attentive statistics pooling + head ----
  gemm(hbuf, poolW1, s1, nullptr, nullptr, nullptr, NTOK, HIDn, DD, DD, HIDn, HIDn,
       0, 0, 0, 0, 1, 1.f, 1.f, /*tanh*/3, 0);
  score_kernel<<<dim3(NTOK / 8), dim3(256), 0, stream>>>(s1, poolW2, scb);
  pool_kernel<<<dim3(BATCH), dim3(512), 0, stream>>>(hbuf, scb, feat);
  head_kernel<<<dim3(BATCH * NCn / 8), dim3(256), 0, stream>>>(feat, headW, headb, out);
}
